// SparseMoELayer_65008624993016
// MI455X (gfx1250) — compile-verified, matrix operand order checked by reading, not run on hardware
//
#include <hip/hip_runtime.h>
#include <hip/hip_bf16.h>
#include <math.h>

typedef _Float16 v16h __attribute__((ext_vector_type(16)));
typedef float    v8f  __attribute__((ext_vector_type(8)));

#define T_TOK   8192
#define DIM_D   1024
#define DIM_H   4096
#define DIM_O   1024
#define NEXP    8
#define MTILE   128
#define NTILE   128
#define KSTEP   32
#define MAX_MT  136          // sum ceil(cnt_e/128) <= 16384/128 + 8
#define MAXROWS 17408        // 16384 + 8*127, rounded up

// ---- workspace layout (bytes) ----
#define WS_COUNTS   0                         // u32[8]
#define WS_CURSORS  32                        // u32[8]
#define WS_PSUM     64                        // f32[8]
#define WS_OFFP     96                        // u32[8]
#define WS_TOKE     256                       // u32[T*2]
#define WS_TOKW     (WS_TOKE + 4*T_TOK*2)     // f32[T*2]
#define WS_ROWS     (WS_TOKW + 4*T_TOK*2)     // u32[MAXROWS]
#define WS_ROWW     (WS_ROWS + 4*MAXROWS)     // f32[MAXROWS]
#define WS_HBUF     (WS_ROWW + 4*MAXROWS)     // _Float16[MAXROWS*DIM_H] (~143 MB)

union H8  { float4 q; _Float16 h[8]; };
union V16 { v16h  v; float4 f[2]; };

// ---------------- kernel 1: zero output + routing header ----------------
__global__ void k_zero(float* __restrict__ out, unsigned int* __restrict__ hdr, int n) {
    int i = blockIdx.x * blockDim.x + threadIdx.x;
    if (i < n) out[i] = 0.0f;
    if (i < 64) hdr[i] = 0u;
}

// ---------------- kernel 2: gating (one wave32 per token) ----------------
__global__ __launch_bounds__(256) void k_gate(
    const float* __restrict__ x, const float* __restrict__ gw,
    const float* __restrict__ gb, unsigned int* __restrict__ cnts,
    float* __restrict__ psum, unsigned int* __restrict__ toke,
    float* __restrict__ tokw) {
    __shared__ float ps[NEXP];
    int tid = threadIdx.x;
    if (tid < NEXP) ps[tid] = 0.0f;
    __syncthreads();

    int t    = blockIdx.x * 8 + (tid >> 5);
    int lane = tid & 31;
    float acc[NEXP];
#pragma unroll
    for (int j = 0; j < NEXP; j++) acc[j] = 0.0f;

    const float* xr = x + (size_t)t * DIM_D;
    for (int i = 0; i < DIM_D / 32; i++) {
        int d = i * 32 + lane;
        float xv = xr[d];
        const float4* g = (const float4*)(gw + (size_t)d * NEXP);
        float4 g0 = g[0], g1 = g[1];
        acc[0] += xv * g0.x; acc[1] += xv * g0.y; acc[2] += xv * g0.z; acc[3] += xv * g0.w;
        acc[4] += xv * g1.x; acc[5] += xv * g1.y; acc[6] += xv * g1.z; acc[7] += xv * g1.w;
    }
#pragma unroll
    for (int j = 0; j < NEXP; j++) {
#pragma unroll
        for (int off = 16; off >= 1; off >>= 1)
            acc[j] += __shfl_xor(acc[j], off, 32);
    }

    if (lane == 0) {
        float s[NEXP], p[NEXP];
        float m = -1e30f;
#pragma unroll
        for (int j = 0; j < NEXP; j++) { s[j] = acc[j] + gb[j]; m = fmaxf(m, s[j]); }
        float sum = 0.0f;
#pragma unroll
        for (int j = 0; j < NEXP; j++) { p[j] = expf(s[j] - m); sum += p[j]; }
        float inv = 1.0f / sum;
#pragma unroll
        for (int j = 0; j < NEXP; j++) { p[j] *= inv; atomicAdd(&ps[j], p[j]); }
        int i0 = 0;
#pragma unroll
        for (int j = 1; j < NEXP; j++) if (p[j] > p[i0]) i0 = j;
        int i1 = (i0 == 0) ? 1 : 0;
#pragma unroll
        for (int j = 0; j < NEXP; j++) if (j != i0 && p[j] > p[i1]) i1 = j;
        float wsum = p[i0] + p[i1];
        toke[2 * t]     = (unsigned int)i0;
        toke[2 * t + 1] = (unsigned int)i1;
        tokw[2 * t]     = p[i0] / wsum;
        tokw[2 * t + 1] = p[i1] / wsum;
        atomicAdd(&cnts[i0], 1u);
        atomicAdd(&cnts[i1], 1u);
    }
    __syncthreads();
    if (tid < NEXP) atomicAdd(&psum[tid], ps[tid]);
}

// ---------------- kernel 3: offsets (padded prefix sum) + aux loss ----------------
__global__ void k_offsets_aux(const unsigned int* __restrict__ cnts,
                              const float* __restrict__ psum,
                              unsigned int* __restrict__ offp,
                              float* __restrict__ aux_out) {
    if (blockIdx.x == 0 && threadIdx.x == 0) {
        unsigned int off = 0;
        for (int e = 0; e < NEXP; e++) {
            offp[e] = off;
            off += ((cnts[e] + MTILE - 1) / MTILE) * MTILE;
        }
        const float invE = 1.0f / (float)NEXP;
        const float logInvE = logf(invE);
        float aux = 0.0f;
        for (int e = 0; e < NEXP; e++) {
            float u = psum[e] / (float)T_TOK;
            aux += u * logInvE - logf(u) * invE;
        }
        *aux_out = aux;
    }
}

// ---------------- kernel 4: scatter token assignments to expert rows ----------------
__global__ void k_scatter(const unsigned int* __restrict__ toke,
                          const float* __restrict__ tokw,
                          const unsigned int* __restrict__ offp,
                          unsigned int* __restrict__ cursors,
                          unsigned int* __restrict__ rows,
                          float* __restrict__ roww) {
    int t = blockIdx.x * blockDim.x + threadIdx.x;
    if (t >= T_TOK) return;
#pragma unroll
    for (int k = 0; k < 2; k++) {
        unsigned int e   = toke[2 * t + k];
        unsigned int pos = offp[e] + atomicAdd(&cursors[e], 1u);
        rows[pos] = (unsigned int)t;
        roww[pos] = tokw[2 * t + k];
    }
}

// Fragment layout (ISA 16-bit A/B 16x32 lane map): element (local, k):
//   lane = ((k>>3)&1)*16 + local ; elem j = (k>>4)*8 + (k&7)
// => for fixed "local" and a 16-wide k slice, the 16 values form two
//    contiguous 8 x f16 (16B) runs -> staged with 2x ds_store_b128.

// ---------------- kernel 5: FFN1  H = gelu(X @ w1[e] + b1[e]) ----------------
__global__ __launch_bounds__(256) void k_ffn1(
    const float* __restrict__ x, const float* __restrict__ w1,
    const float* __restrict__ b1, const unsigned int* __restrict__ cnts,
    const unsigned int* __restrict__ offp, const unsigned int* __restrict__ rows,
    _Float16* __restrict__ hbuf) {
    __shared__ alignas(16) _Float16 lA[2 * 8 * 32 * 16];   // double-buffered fragment tiles
    __shared__ alignas(16) _Float16 lB[2 * 8 * 32 * 16];

    // map blockIdx.x -> (expert, m-tile)
    int bx = blockIdx.x, e = -1, mt = 0, tb = 0;
    for (int i = 0; i < NEXP; i++) {
        int tc = (int)((cnts[i] + MTILE - 1) / MTILE);
        if (bx < tb + tc) { e = i; mt = bx - tb; break; }
        tb += tc;
    }
    if (e < 0) return;
    int cnt       = (int)cnts[e];
    int rowsLocal = cnt - mt * MTILE; if (rowsLocal > MTILE) rowsLocal = MTILE;
    int base      = (int)offp[e] + mt * MTILE;
    int nbase     = blockIdx.y * NTILE;
    int tid       = threadIdx.x;

    // A staging: thread = (row ar, 16-wide k slice akh); fp32 x gathered by token.
    // Padded rows only need a SAFE address (their C rows are never stored), so
    // clamp instead of branching -> straight-line staging, no exec-mask churn.
    int ar  = tid & 127, akh = tid >> 7;
    int arc = ar < rowsLocal ? ar : (rowsLocal - 1);   // tile exists => rowsLocal >= 1
    int atok = (int)rows[base + arc];
    const float* aptr = x + (size_t)atok * DIM_D + akh * 16;
    int s_a = ar >> 4, rl = ar & 15;
    int aIdx0 = (s_a << 9) + (rl << 4) + (akh << 3);
    int aIdx1 = (s_a << 9) + ((16 + rl) << 4) + (akh << 3);

    // B staging: thread = (n column nl, 16-wide k slice kh); column loads, coalesced over nl
    int nl = tid & 127, kh = tid >> 7;
    const float* bcol = w1 + (size_t)e * DIM_D * DIM_H + (size_t)(kh * 16) * DIM_H + nbase + nl;
    int sn = nl >> 4;
    int bIdx0 = (sn << 9) + ((nl & 15) << 4) + (kh << 3);
    int bIdx1 = (sn << 9) + ((16 + (nl & 15)) << 4) + (kh << 3);

    int lane = tid & 31, wv = tid >> 5;
    int mg = wv >> 2, ng = wv & 3;

    v8f zero = {0.f, 0.f, 0.f, 0.f, 0.f, 0.f, 0.f, 0.f};
    v8f acc[4][2];
#pragma unroll
    for (int a = 0; a < 4; a++)
#pragma unroll
        for (int b = 0; b < 2; b++) acc[a][b] = zero;

    auto stage = [&](int kb, int buf) {
        const float4* p = (const float4*)(aptr + kb);
        float4 q0 = p[0], q1 = p[1], q2 = p[2], q3 = p[3];
        H8 a0, a1;
        a0.h[0]=(_Float16)q0.x; a0.h[1]=(_Float16)q0.y; a0.h[2]=(_Float16)q0.z; a0.h[3]=(_Float16)q0.w;
        a0.h[4]=(_Float16)q1.x; a0.h[5]=(_Float16)q1.y; a0.h[6]=(_Float16)q1.z; a0.h[7]=(_Float16)q1.w;
        a1.h[0]=(_Float16)q2.x; a1.h[1]=(_Float16)q2.y; a1.h[2]=(_Float16)q2.z; a1.h[3]=(_Float16)q2.w;
        a1.h[4]=(_Float16)q3.x; a1.h[5]=(_Float16)q3.y; a1.h[6]=(_Float16)q3.z; a1.h[7]=(_Float16)q3.w;
        _Float16* A = lA + (buf << 12);
        *(float4*)&A[aIdx0] = a0.q;
        *(float4*)&A[aIdx1] = a1.q;

        const float* pb = bcol + (size_t)kb * DIM_H;
        float f[16];
#pragma unroll
        for (int kk = 0; kk < 16; kk++) f[kk] = pb[(size_t)kk * DIM_H];
        H8 b0, b1;
#pragma unroll
        for (int i = 0; i < 8; i++) { b0.h[i] = (_Float16)f[i]; b1.h[i] = (_Float16)f[8 + i]; }
        _Float16* B = lB + (buf << 12);
        *(float4*)&B[bIdx0] = b0.q;
        *(float4*)&B[bIdx1] = b1.q;
    };

    auto compute = [&](int buf) {
        const _Float16* A = lA + (buf << 12);
        const _Float16* B = lB + (buf << 12);
        v16h va[4], vb[2];
#pragma unroll
        for (int a = 0; a < 4; a++)
            va[a] = *(const v16h*)&A[(((mg << 2) + a) << 9) + (lane << 4)];
#pragma unroll
        for (int b = 0; b < 2; b++)
            vb[b] = *(const v16h*)&B[(((ng << 1) + b) << 9) + (lane << 4)];
#pragma unroll
        for (int a = 0; a < 4; a++)
#pragma unroll
            for (int b = 0; b < 2; b++)
                acc[a][b] = __builtin_amdgcn_wmma_f32_16x16x32_f16(
                    false, va[a], false, vb[b], (short)0, acc[a][b], false, false);
    };

    const int NSTEP = DIM_D / KSTEP;
    stage(0, 0);
    __syncthreads();
    for (int it = 0; it < NSTEP; ++it) {
        if (it + 1 < NSTEP) stage((it + 1) * KSTEP, (it + 1) & 1);
        compute(it & 1);
        __syncthreads();
    }

    // epilogue: bias + exact gelu, store f16
    int rl8 = (lane >> 4) * 8, nlc = lane & 15;
#pragma unroll
    for (int a = 0; a < 4; a++) {
        int s = (mg << 2) + a;
#pragma unroll
        for (int b = 0; b < 2; b++) {
            int col = nbase + ((ng << 1) + b) * 16 + nlc;
            float bias = b1[e * DIM_H + col];
#pragma unroll
            for (int r = 0; r < 8; r++) {
                int rowT = s * 16 + rl8 + r;
                if (rowT < rowsLocal) {
                    float v = acc[a][b][r] + bias;
                    v = 0.5f * v * (1.0f + erff(v * 0.70710678118f));
                    hbuf[(size_t)(base + rowT) * DIM_H + col] = (_Float16)v;
                }
            }
        }
    }
}

// ---------------- kernel 6: FFN2  out += roww * (H @ w2[e] + b2[e]) ----------------
__global__ __launch_bounds__(256) void k_ffn2(
    const float* __restrict__ w2, const float* __restrict__ b2,
    const unsigned int* __restrict__ cnts, const unsigned int* __restrict__ offp,
    const unsigned int* __restrict__ rows, const float* __restrict__ roww,
    const _Float16* __restrict__ hbuf, float* __restrict__ out) {
    __shared__ alignas(16) _Float16 lB[2 * 8 * 32 * 16];   // B only; A loads direct (f16 row-major)

    int bx = blockIdx.x, e = -1, mt = 0, tb = 0;
    for (int i = 0; i < NEXP; i++) {
        int tc = (int)((cnts[i] + MTILE - 1) / MTILE);
        if (bx < tb + tc) { e = i; mt = bx - tb; break; }
        tb += tc;
    }
    if (e < 0) return;
    int cnt       = (int)cnts[e];
    int rowsLocal = cnt - mt * MTILE; if (rowsLocal > MTILE) rowsLocal = MTILE;
    int base      = (int)offp[e] + mt * MTILE;
    int nbase     = blockIdx.y * NTILE;
    int tid       = threadIdx.x;

    int nl = tid & 127, kh = tid >> 7;
    const float* bcol = w2 + (size_t)e * DIM_H * DIM_O + (size_t)(kh * 16) * DIM_O + nbase + nl;
    int sn = nl >> 4;
    int bIdx0 = (sn << 9) + ((nl & 15) << 4) + (kh << 3);
    int bIdx1 = (sn << 9) + ((16 + (nl & 15)) << 4) + (kh << 3);

    int lane = tid & 31, wv = tid >> 5;
    int mg = wv >> 2, ng = wv & 3;

    // per-lane direct A fragment pointers (row-major f16 == A fragment layout, 2x b128/lane)
    const _Float16* apA[4];
#pragma unroll
    for (int a = 0; a < 4; a++) {
        int rowT = ((mg << 2) + a) * 16 + (lane & 15);
        apA[a] = hbuf + (size_t)(base + rowT) * DIM_H + ((lane >> 4) << 3);  // padded rows: safe garbage
    }

    v8f zero = {0.f, 0.f, 0.f, 0.f, 0.f, 0.f, 0.f, 0.f};
    v8f acc[4][2];
#pragma unroll
    for (int a = 0; a < 4; a++)
#pragma unroll
        for (int b = 0; b < 2; b++) acc[a][b] = zero;

    auto stageB = [&](int kb, int buf) {
        const float* pb = bcol + (size_t)kb * DIM_O;
        float f[16];
#pragma unroll
        for (int kk = 0; kk < 16; kk++) f[kk] = pb[(size_t)kk * DIM_O];
        H8 b0, b1;
#pragma unroll
        for (int i = 0; i < 8; i++) { b0.h[i] = (_Float16)f[i]; b1.h[i] = (_Float16)f[8 + i]; }
        _Float16* B = lB + (buf << 12);
        *(float4*)&B[bIdx0] = b0.q;
        *(float4*)&B[bIdx1] = b1.q;
    };

    auto compute = [&](int kb, int buf) {
        const _Float16* B = lB + (buf << 12);
        v16h va[4], vb[2];
#pragma unroll
        for (int a = 0; a < 4; a++) {
            V16 u;
            u.f[0] = *(const float4*)(apA[a] + kb);
            u.f[1] = *(const float4*)(apA[a] + kb + 16);
            va[a] = u.v;
        }
#pragma unroll
        for (int b = 0; b < 2; b++)
            vb[b] = *(const v16h*)&B[(((ng << 1) + b) << 9) + (lane << 4)];
#pragma unroll
        for (int a = 0; a < 4; a++)
#pragma unroll
            for (int b = 0; b < 2; b++)
                acc[a][b] = __builtin_amdgcn_wmma_f32_16x16x32_f16(
                    false, va[a], false, vb[b], (short)0, acc[a][b], false, false);
    };

    const int NSTEP = DIM_H / KSTEP;
    stageB(0, 0);
    __syncthreads();
    for (int it = 0; it < NSTEP; ++it) {
        if (it + 1 < NSTEP) stageB((it + 1) * KSTEP, (it + 1) & 1);
        compute(it * KSTEP, it & 1);
        __syncthreads();
    }

    // epilogue: bias + gate weight, atomic combine into out (2 adds/elem -> deterministic)
    int rl8 = (lane >> 4) * 8, nlc = lane & 15;
#pragma unroll
    for (int a = 0; a < 4; a++) {
        int s = (mg << 2) + a;
#pragma unroll
        for (int b = 0; b < 2; b++) {
            int col = nbase + ((ng << 1) + b) * 16 + nlc;
            float bias = b2[e * DIM_O + col];
#pragma unroll
            for (int r = 0; r < 8; r++) {
                int rowT = s * 16 + rl8 + r;
                if (rowT < rowsLocal) {
                    int gRow = base + rowT;
                    int t = (int)rows[gRow];
                    float v = (acc[a][b][r] + bias) * roww[gRow];
                    atomicAdd(&out[(size_t)t * DIM_O + col], v);
                }
            }
        }
    }
}

// ---------------- launcher ----------------
extern "C" void kernel_launch(void* const* d_in, const int* in_sizes, int n_in,
                              void* d_out, int out_size, void* d_ws, size_t ws_size,
                              hipStream_t stream) {
    const float* x  = (const float*)d_in[0];
    const float* gw = (const float*)d_in[1];
    const float* gb = (const float*)d_in[2];
    const float* w1 = (const float*)d_in[3];
    const float* b1 = (const float*)d_in[4];
    const float* w2 = (const float*)d_in[5];
    const float* b2 = (const float*)d_in[6];
    float* out = (float*)d_out;

    char* ws = (char*)d_ws;
    unsigned int* cnts    = (unsigned int*)(ws + WS_COUNTS);
    unsigned int* cursors = (unsigned int*)(ws + WS_CURSORS);
    float*        psum    = (float*)(ws + WS_PSUM);
    unsigned int* offp    = (unsigned int*)(ws + WS_OFFP);
    unsigned int* toke    = (unsigned int*)(ws + WS_TOKE);
    float*        tokw    = (float*)(ws + WS_TOKW);
    unsigned int* rows    = (unsigned int*)(ws + WS_ROWS);
    float*        roww    = (float*)(ws + WS_ROWW);
    _Float16*     hbuf    = (_Float16*)(ws + WS_HBUF);

    k_zero<<<(out_size + 255) / 256, 256, 0, stream>>>(out, (unsigned int*)ws, out_size);
    k_gate<<<T_TOK / 8, 256, 0, stream>>>(x, gw, gb, cnts, psum, toke, tokw);
    k_offsets_aux<<<1, 32, 0, stream>>>(cnts, psum, offp, out + (out_size - 1));
    k_scatter<<<T_TOK / 256, 256, 0, stream>>>(toke, tokw, offp, cursors, rows, roww);
    k_ffn1<<<dim3(MAX_MT, DIM_H / NTILE), 256, 0, stream>>>(x, w1, b1, cnts, offp, rows, hbuf);
    k_ffn2<<<dim3(MAX_MT, DIM_O / NTILE), 256, 0, stream>>>(w2, b2, cnts, offp, rows, roww, hbuf, out);
}
